// CrossAttention_37383395345085
// MI455X (gfx1250) — matrix operand order repaired and checked by reading, not compiled, not run on hardware
//
#include <hip/hip_runtime.h>

#define DEVI __device__ __forceinline__

typedef __attribute__((ext_vector_type(16))) __bf16 v16bf;
typedef __attribute__((ext_vector_type(8)))  float  v8f;

DEVI unsigned short f2bf(float f) {
  unsigned u = __float_as_uint(f);
  return (unsigned short)((u + 0x7fffu + ((u >> 16) & 1u)) >> 16);
}

union Frag16 { int i[8]; unsigned short u[16]; v16bf v; };
union FragAcc { float f[8]; v8f v; };

DEVI v8f wmma_bf16(v16bf a, v16bf b, v8f c) {
  // D = A(16x32 bf16) * B(32x16 bf16) + C(16x16 f32)
  return __builtin_amdgcn_wmma_f32_16x16x32_bf16(false, a, false, b, (short)0, c,
                                                 false, false);
}

// A-fragment (16x32 bf16, row-major source, stride in elements).
// Lane L<16: M=L, K 0..15 contiguous
// Lane L>=16: M=L-16, K 16..31 contiguous.  Contiguous -> merges to b128s.
DEVI v16bf load_a_frag(const unsigned short* base, int stride) {
  int lane = threadIdx.x & 31;
  int r    = lane & 15;
  int hi   = lane >> 4;
  Frag16 f;
#pragma unroll
  for (int v = 0; v < 8; ++v) {
    int kb = 2 * v + (hi ? 16 : 0);
    f.i[v] = *(const int*)(base + r * stride + kb);
  }
  return f.v;
}

// B-fragment (32x16 bf16) from a tile stored TRANSPOSED [N][K] (K contiguous),
// column stride cs. Lane: N = L%16, K-half = (L>=16)?16:0.
// All 8 dwords per lane are contiguous -> merges to 2x ds_load_b128.
DEVI v16bf load_bt_frag(const unsigned short* base, int cs) {
  int lane = threadIdx.x & 31;
  int c    = lane & 15;
  int kh   = (lane >> 4) * 16;
  Frag16 f;
#pragma unroll
  for (int v = 0; v < 8; ++v)
    f.i[v] = *(const int*)(base + c * cs + kh + 2 * v);
  return f.v;
}

// ---------------------------------------------------------------- convert ---
__global__ void cvt_bf16_kern(const float* __restrict__ in,
                              unsigned short* __restrict__ out, int n) {
  int i = blockIdx.x * blockDim.x + threadIdx.x;
  if (i < n) out[i] = f2bf(in[i]);
}

// W[co][ci][kk] (kk=kh*3+kw) -> out[kk][co][ci], bf16
__global__ void repack_w3_kern(const float* __restrict__ w,
                               unsigned short* __restrict__ out) {
  int i = blockIdx.x * blockDim.x + threadIdx.x;
  if (i < 512 * 512 * 9) {
    int kk = i % 9;
    int r  = i / 9;
    int ci = r & 511;
    int co = r >> 9;
    out[(kk * 512 + co) * 512 + ci] = f2bf(w[i]);
  }
}

// ------------------------------------------------------------------- conv ---
// implicit-GEMM 3x3 conv, pad 1:  Y[n][co][p] = sum_{kk,ci} Wt[kk][co][ci] *
// X[n][ci][ph+dy][pw+dx].  Block tile 128(M=co) x 128(N=p), 8 waves (4x2).
// Double-buffered LDS with register staging: one barrier per K-tile; the next
// tile's global loads are in flight while the current tile's WMMAs issue.
__global__ __launch_bounds__(256)
void conv3x3_wmma(const unsigned short* __restrict__ X,   // [4][512][32][32] bf16
                  const unsigned short* __restrict__ Wt,  // [9][512][512]    bf16
                  unsigned short* __restrict__ Y)         // [4][512][1024]   bf16
{
  __shared__ __align__(16) unsigned short Ash[2][128 * 32];  // [co][ci]
  __shared__ __align__(16) unsigned short Bsh[2][128 * 32];  // TRANSPOSED [p][ci]
  const int t  = threadIdx.x;
  const int m0 = blockIdx.x * 128;
  const int p0 = blockIdx.y * 128;
  const int n  = blockIdx.z;
  const int wv = t >> 5;
  const int wm = wv & 3;   // M sub-tile 32*wm
  const int wn = wv >> 2;  // N sub-tile 64*wn

  FragAcc acc[2][4];
#pragma unroll
  for (int i = 0; i < 2; ++i)
#pragma unroll
    for (int j = 0; j < 4; ++j)
#pragma unroll
      for (int e = 0; e < 8; ++e) acc[i][j].f[e] = 0.f;

  // register staging for the in-flight tile
  uint4 areg0, areg1;
  unsigned short breg[16];

  const int arow = t >> 1, acblk = (t & 1) * 16;   // A staging coords
  const int brow = t >> 3, bseg = (t & 7) * 16;    // B staging coords (ci, p)

  auto load_tile = [&](int iter) {  // tile iter: kk = iter/16, cb = iter%16
    int kk = iter >> 4, cb = iter & 15;
    int dy = kk / 3 - 1, dx = kk % 3 - 1;
    int c0 = cb * 32;
    const unsigned short* Wk = Wt + (size_t)kk * 512 * 512;
    {
      const uint4* src = (const uint4*)(Wk + (size_t)(m0 + arow) * 512 + c0 + acblk);
      areg0 = src[0];
      areg1 = src[1];
      if (iter + 1 < 144)  // keep the weight stream one more tile ahead in L2
        __builtin_prefetch((const void*)(Wk + (size_t)(m0 + arow) * 512 + c0 + 32 + acblk), 0, 3);
    }
    {
      int p  = p0 + bseg;
      int ph = p >> 5, w0 = p & 31;
      int hh = ph + dy;
      bool rowok = ((unsigned)hh < 32u);
      const unsigned short* xr =
          X + ((size_t)(n * 512 + c0 + brow) * 32 + (rowok ? hh : 0)) * 32;
#pragma unroll
      for (int j = 0; j < 16; ++j) {
        int ww = w0 + dx + j;
        breg[j] = (rowok && (unsigned)ww < 32u) ? xr[ww] : (unsigned short)0;
      }
    }
  };
  auto store_tile = [&](int buf) {
    uint4* dst = (uint4*)(Ash[buf] + arow * 32 + acblk);
    dst[0] = areg0;
    dst[1] = areg1;
#pragma unroll
    for (int j = 0; j < 16; ++j) Bsh[buf][(bseg + j) * 32 + brow] = breg[j];
  };

  load_tile(0);
  store_tile(0);
  __syncthreads();

  for (int iter = 0; iter < 144; ++iter) {
    const int cur = iter & 1;
    if (iter + 1 < 144) load_tile(iter + 1);  // global loads overlap compute

    v16bf a[2], b[4];
#pragma unroll
    for (int i = 0; i < 2; ++i)
      a[i] = load_a_frag(Ash[cur] + (wm * 32 + i * 16) * 32, 32);
#pragma unroll
    for (int j = 0; j < 4; ++j)
      b[j] = load_bt_frag(Bsh[cur] + (wn * 64 + j * 16) * 32, 32);
#pragma unroll
    for (int i = 0; i < 2; ++i)
#pragma unroll
      for (int j = 0; j < 4; ++j) acc[i][j].v = wmma_bf16(a[i], b[j], acc[i][j].v);

    if (iter + 1 < 144) store_tile(1 - cur);  // buffer last read before prev barrier
    __syncthreads();
  }

  // C layout: lane L: col = L%16, row = v + 8*(L>=16)
  int lane = t & 31, cc = lane & 15, rofs = (lane >> 4) * 8;
#pragma unroll
  for (int i = 0; i < 2; ++i)
#pragma unroll
    for (int j = 0; j < 4; ++j)
#pragma unroll
      for (int v = 0; v < 8; ++v) {
        int row = m0 + wm * 32 + i * 16 + rofs + v;
        int col = p0 + wn * 64 + j * 16 + cc;
        Y[(size_t)(n * 512 + row) * 1024 + col] = f2bf(acc[i][j].f[v]);
      }
}

// -------------------------------------------------------------- attention ---
// flash-style: per (n, head, q-block of 128 rows). q,k,v bf16 in [n][512][1024]
// (channel-major; head h owns spatial slice [h*128, h*128+128)).  Each wave
// owns 16 q rows.  K staged [c_k][d]; V staged TRANSPOSED [d][c_k] (same buf).
__global__ __launch_bounds__(256)
void attn_wmma(const unsigned short* __restrict__ Q,
               const unsigned short* __restrict__ K,
               const unsigned short* __restrict__ V,
               unsigned short* __restrict__ Y)
{
  __shared__ __align__(16) unsigned short KVsh[64 * 128];   // 16KB, K then V^T
  __shared__ __align__(16) unsigned short Psh[8 * 16 * 64]; // 16KB, wave-private P
  const int t    = threadIdx.x;
  const int lane = t & 31;
  const int wv   = t >> 5;
  const int q0   = blockIdx.x * 128;
  const int head = blockIdx.y;
  const int n    = blockIdx.z;
  const int qr0  = q0 + wv * 16;
  const float scale = 0.08838834764831845f;  // 1/sqrt(128)

  // q A-frags (16 rows x 128 d) straight from global into registers
  v16bf qa[4];
  {
    const unsigned short* qb = Q + (size_t)(n * 512 + qr0) * 1024 + head * 128;
#pragma unroll
    for (int ks = 0; ks < 4; ++ks) qa[ks] = load_a_frag(qb + ks * 32, 1024);
  }

  FragAcc o[8];
#pragma unroll
  for (int j = 0; j < 8; ++j)
#pragma unroll
    for (int e = 0; e < 8; ++e) o[j].f[e] = 0.f;
  float mrun[8], lrun[8];
#pragma unroll
  for (int v = 0; v < 8; ++v) { mrun[v] = -1e30f; lrun[v] = 0.f; }

  for (int kb = 0; kb < 8; ++kb) {
    const int kb0 = kb * 64;
    __syncthreads();
    {  // stage K block [64 c_k][128 d] (natural layout, b128 in and out)
      int row = t >> 2, cbase = (t & 3) * 32;
      const uint4* src =
          (const uint4*)(K + (size_t)(n * 512 + kb0 + row) * 1024 + head * 128 + cbase);
      uint4* dst = (uint4*)(KVsh + row * 128 + cbase);
#pragma unroll
      for (int j = 0; j < 4; ++j) dst[j] = src[j];
    }
    __syncthreads();

    // S = q * k^T (contract over d=128); K tile is [c_k][d] -> bt frags
    FragAcc s[4];
#pragma unroll
    for (int j = 0; j < 4; ++j)
#pragma unroll
      for (int e = 0; e < 8; ++e) s[j].f[e] = 0.f;
#pragma unroll
    for (int ks = 0; ks < 4; ++ks)
#pragma unroll
      for (int j = 0; j < 4; ++j) {
        v16bf b = load_bt_frag(KVsh + (j * 16) * 128 + ks * 32, 128);
        s[j].v  = wmma_bf16(qa[ks], b, s[j].v);
      }

    // online softmax (row = VGPR index v, cols across 16 lanes of a half-wave)
#pragma unroll
    for (int v = 0; v < 8; ++v) {
      float mx = -1e30f;
#pragma unroll
      for (int j = 0; j < 4; ++j) mx = fmaxf(mx, s[j].f[v] * scale);
      mx = fmaxf(mx, __shfl_xor(mx, 1));
      mx = fmaxf(mx, __shfl_xor(mx, 2));
      mx = fmaxf(mx, __shfl_xor(mx, 4));
      mx = fmaxf(mx, __shfl_xor(mx, 8));
      float mnew = fmaxf(mrun[v], mx);
      float sum  = 0.f;
#pragma unroll
      for (int j = 0; j < 4; ++j) {
        float p   = __expf(s[j].f[v] * scale - mnew);
        s[j].f[v] = p;
        sum += p;
      }
      sum += __shfl_xor(sum, 1);
      sum += __shfl_xor(sum, 2);
      sum += __shfl_xor(sum, 4);
      sum += __shfl_xor(sum, 8);
      float corr = __expf(mrun[v] - mnew);
      lrun[v]    = lrun[v] * corr + sum;
      mrun[v]    = mnew;
#pragma unroll
      for (int j = 0; j < 8; ++j) o[j].f[v] *= corr;
    }

    // P -> wave-private LDS as [16][64] (re-layout C-frag -> A-frag source)
    {
      unsigned short* pw = Psh + wv * 16 * 64;
      int cc = lane & 15, hi = lane >> 4;
#pragma unroll
      for (int j = 0; j < 4; ++j)
#pragma unroll
        for (int v = 0; v < 8; ++v)
          pw[(v + hi * 8) * 64 + j * 16 + cc] = f2bf(s[j].f[v]);
    }

    __syncthreads();  // all waves done reading K
    {  // stage V block TRANSPOSED: Vt[d=128][c_k=64] into the same buffer
      int row = t >> 2, cbase = (t & 3) * 32;  // row = c_k, cbase = d offset
      const uint4* src =
          (const uint4*)(V + (size_t)(n * 512 + kb0 + row) * 1024 + head * 128 + cbase);
      union { uint4 q[4]; unsigned short h[32]; } u;
#pragma unroll
      for (int j = 0; j < 4; ++j) u.q[j] = src[j];
#pragma unroll
      for (int j = 0; j < 32; ++j) KVsh[(cbase + j) * 64 + row] = u.h[j];
    }
    __syncthreads();

    // O += P * V  (contract over c_k = 64 -> 2 WMMA k-steps); V^T -> bt frags
    const unsigned short* pw = Psh + wv * 16 * 64;
#pragma unroll
    for (int ks = 0; ks < 2; ++ks) {
      v16bf pa = load_a_frag(pw + ks * 32, 64);
#pragma unroll
      for (int j = 0; j < 8; ++j) {
        v16bf b = load_bt_frag(KVsh + (j * 16) * 64 + ks * 32, 64);
        o[j].v  = wmma_bf16(pa, b, o[j].v);
      }
    }
  }

  // normalize + store bf16 into [n][512][1024] (head slice)
  {
    int cc = lane & 15, hi = lane >> 4;
#pragma unroll
    for (int v = 0; v < 8; ++v) {
      float inv = 1.0f / lrun[v];
      int row   = qr0 + v + hi * 8;
#pragma unroll
      for (int j = 0; j < 8; ++j)
        Y[(size_t)(n * 512 + row) * 1024 + head * 128 + j * 16 + cc] =
            f2bf(o[j].f[v] * inv);
    }
  }
}

// ---------------------------------------------------- 1x1 proj + residual ---
// Double-buffered like the conv kernel.
__global__ __launch_bounds__(256)
void proj_res_wmma(const unsigned short* __restrict__ Ybf,  // [4][512][1024] bf16
                   const unsigned short* __restrict__ Wp,   // [512][512]     bf16
                   const float* __restrict__ Xres,          // [4][512][1024] f32
                   const float* __restrict__ rwp,           // resweight
                   float* __restrict__ Out)
{
  __shared__ __align__(16) unsigned short Ash[2][128 * 32];  // [co][ci]
  __shared__ __align__(16) unsigned short Bsh[2][128 * 32];  // TRANSPOSED [p][ci]
  const int t  = threadIdx.x;
  const int m0 = blockIdx.x * 128, p0 = blockIdx.y * 128, n = blockIdx.z;
  const int wv = t >> 5, wm = wv & 3, wn = wv >> 2;
  const float rw = rwp[0];

  FragAcc acc[2][4];
#pragma unroll
  for (int i = 0; i < 2; ++i)
#pragma unroll
    for (int j = 0; j < 4; ++j)
#pragma unroll
      for (int e = 0; e < 8; ++e) acc[i][j].f[e] = 0.f;

  uint4 areg0, areg1, breg0, breg1;
  const int arow = t >> 1, acblk = (t & 1) * 16;
  const int brow = t >> 3, bseg = (t & 7) * 16;

  auto load_tile = [&](int cb) {
    int c0 = cb * 32;
    const uint4* asrc = (const uint4*)(Wp + (size_t)(m0 + arow) * 512 + c0 + acblk);
    areg0 = asrc[0];
    areg1 = asrc[1];
    const uint4* bsrc =
        (const uint4*)(Ybf + (size_t)(n * 512 + c0 + brow) * 1024 + p0 + bseg);
    breg0 = bsrc[0];
    breg1 = bsrc[1];
  };
  auto store_tile = [&](int buf) {
    uint4* dst = (uint4*)(Ash[buf] + arow * 32 + acblk);
    dst[0] = areg0;
    dst[1] = areg1;
    union { uint4 q[2]; unsigned short h[16]; } u;
    u.q[0] = breg0;
    u.q[1] = breg1;
#pragma unroll
    for (int j = 0; j < 16; ++j) Bsh[buf][(bseg + j) * 32 + brow] = u.h[j];
  };

  load_tile(0);
  store_tile(0);
  __syncthreads();

  for (int cb = 0; cb < 16; ++cb) {
    const int cur = cb & 1;
    if (cb + 1 < 16) load_tile(cb + 1);

    v16bf a[2], b[4];
#pragma unroll
    for (int i = 0; i < 2; ++i)
      a[i] = load_a_frag(Ash[cur] + (wm * 32 + i * 16) * 32, 32);
#pragma unroll
    for (int j = 0; j < 4; ++j)
      b[j] = load_bt_frag(Bsh[cur] + (wn * 64 + j * 16) * 32, 32);
#pragma unroll
    for (int i = 0; i < 2; ++i)
#pragma unroll
      for (int j = 0; j < 4; ++j) acc[i][j].v = wmma_bf16(a[i], b[j], acc[i][j].v);

    if (cb + 1 < 16) store_tile(1 - cur);
    __syncthreads();
  }

  int lane = t & 31, cc = lane & 15, rofs = (lane >> 4) * 8;
#pragma unroll
  for (int i = 0; i < 2; ++i)
#pragma unroll
    for (int j = 0; j < 4; ++j)
#pragma unroll
      for (int v = 0; v < 8; ++v) {
        int row    = m0 + wm * 32 + i * 16 + rofs + v;
        int col    = p0 + wn * 64 + j * 16 + cc;
        size_t idx = (size_t)(n * 512 + row) * 1024 + col;
        Out[idx]   = Xres[idx] + acc[i][j].f[v] * rw;
      }
}

// ---------------------------------------------------------------- launch ----
extern "C" void kernel_launch(void* const* d_in, const int* in_sizes, int n_in,
                              void* d_out, int out_size, void* d_ws, size_t ws_size,
                              hipStream_t stream) {
  (void)in_sizes; (void)n_in; (void)out_size; (void)ws_size;

  const float* x_l = (const float*)d_in[0];
  const float* x_g = (const float*)d_in[1];
  const float* Wk1 = (const float*)d_in[2];
  const float* Wq1 = (const float*)d_in[3];
  const float* Wv1 = (const float*)d_in[4];
  const float* Wk2 = (const float*)d_in[5];
  const float* Wq2 = (const float*)d_in[6];
  const float* Wv2 = (const float*)d_in[7];
  const float* Wp1 = (const float*)d_in[8];
  const float* Wp2 = (const float*)d_in[9];
  const float* rw  = (const float*)d_in[10];

  const size_t NX  = (size_t)4 * 512 * 1024;  // activation elems
  const size_t NW3 = (size_t)512 * 512 * 9;
  const size_t NW1 = (size_t)512 * 512;

  char* ws   = (char*)d_ws;
  size_t off = 0;
  auto alloc = [&](size_t bytes) {
    void* p = ws + off;
    off     = (off + bytes + 255) & ~(size_t)255;
    return p;
  };
  unsigned short* xl_b  = (unsigned short*)alloc(NX * 2);
  unsigned short* xg_b  = (unsigned short*)alloc(NX * 2);
  unsigned short* wk1_b = (unsigned short*)alloc(NW3 * 2);
  unsigned short* wq1_b = (unsigned short*)alloc(NW3 * 2);
  unsigned short* wv1_b = (unsigned short*)alloc(NW3 * 2);
  unsigned short* wk2_b = (unsigned short*)alloc(NW3 * 2);
  unsigned short* wq2_b = (unsigned short*)alloc(NW3 * 2);
  unsigned short* wv2_b = (unsigned short*)alloc(NW3 * 2);
  unsigned short* wp1_b = (unsigned short*)alloc(NW1 * 2);
  unsigned short* wp2_b = (unsigned short*)alloc(NW1 * 2);
  unsigned short* kl_b  = (unsigned short*)alloc(NX * 2);
  unsigned short* ql_b  = (unsigned short*)alloc(NX * 2);
  unsigned short* vl_b  = (unsigned short*)alloc(NX * 2);
  unsigned short* kg_b  = (unsigned short*)alloc(NX * 2);
  unsigned short* qg_b  = (unsigned short*)alloc(NX * 2);
  unsigned short* vg_b  = (unsigned short*)alloc(NX * 2);
  unsigned short* yl_b  = (unsigned short*)alloc(NX * 2);
  unsigned short* yg_b  = (unsigned short*)alloc(NX * 2);

  dim3 blk(256);
  int gx  = (int)((NX + 255) / 256);
  int gw3 = (int)((NW3 + 255) / 256);
  int gw1 = (int)((NW1 + 255) / 256);

  cvt_bf16_kern<<<gx, blk, 0, stream>>>(x_l, xl_b, (int)NX);
  cvt_bf16_kern<<<gx, blk, 0, stream>>>(x_g, xg_b, (int)NX);
  repack_w3_kern<<<gw3, blk, 0, stream>>>(Wk1, wk1_b);
  repack_w3_kern<<<gw3, blk, 0, stream>>>(Wq1, wq1_b);
  repack_w3_kern<<<gw3, blk, 0, stream>>>(Wv1, wv1_b);
  repack_w3_kern<<<gw3, blk, 0, stream>>>(Wk2, wk2_b);
  repack_w3_kern<<<gw3, blk, 0, stream>>>(Wq2, wq2_b);
  repack_w3_kern<<<gw3, blk, 0, stream>>>(Wv2, wv2_b);
  cvt_bf16_kern<<<gw1, blk, 0, stream>>>(Wp1, wp1_b, (int)NW1);
  cvt_bf16_kern<<<gw1, blk, 0, stream>>>(Wp2, wp2_b, (int)NW1);

  dim3 grid(4, 8, 4);  // (M/128, N(HW)/128, batch) — also (qblk, head, batch)
  conv3x3_wmma<<<grid, blk, 0, stream>>>(xl_b, wk1_b, kl_b);
  conv3x3_wmma<<<grid, blk, 0, stream>>>(xl_b, wq1_b, ql_b);
  conv3x3_wmma<<<grid, blk, 0, stream>>>(xl_b, wv1_b, vl_b);
  conv3x3_wmma<<<grid, blk, 0, stream>>>(xg_b, wk2_b, kg_b);
  conv3x3_wmma<<<grid, blk, 0, stream>>>(xg_b, wq2_b, qg_b);
  conv3x3_wmma<<<grid, blk, 0, stream>>>(xg_b, wv2_b, vg_b);

  attn_wmma<<<grid, blk, 0, stream>>>(qg_b, kl_b, vl_b, yg_b);  // y_g
  attn_wmma<<<grid, blk, 0, stream>>>(ql_b, kg_b, vg_b, yl_b);  // y_l

  float* out_l = (float*)d_out;
  float* out_g = out_l + NX;
  proj_res_wmma<<<grid, blk, 0, stream>>>(yl_b, wp1_b, x_l, rw, out_l);
  proj_res_wmma<<<grid, blk, 0, stream>>>(yg_b, wp2_b, x_g, rw, out_g);
}
